// RelationNet_85916525789873
// MI455X (gfx1250) — matrix operand order repaired and checked by reading, not compiled, not run on hardware
//
#include <hip/hip_runtime.h>
#include <math.h>

// ---------------------------------------------------------------------------
// RelationNet pipeline for MI455X (gfx1250), wave32 + WMMA f16 16x16x32.
// BS=8, N=900, D=256, K=10.  All GEMMs run on v_wmma_f32_16x16x32_f16 with
// 16x64 output tiles per wave (A-fragment reused across 4 WMMAs).
// ---------------------------------------------------------------------------

typedef __attribute__((ext_vector_type(16))) _Float16 v16h;
typedef __attribute__((ext_vector_type(8)))  _Float16 v8h;
typedef __attribute__((ext_vector_type(8)))  float    v8f;

#define BSN   7200      // BS*N rows
#define NN    900
#define DD    256
#define KK    10
#define FEATC 576       // 64 + 512
#define FROWS 72000     // BSN*KK
#define CH_R  8000      // feature rows per chunk (multiple of 16), 9 chunks
#define CH_BI 800       // (b,i) rows per chunk

// ---------------- f32 -> f16 helpers ---------------------------------------

__global__ void cvt_f16_kernel(const float* __restrict__ src,
                               _Float16* __restrict__ dst, int n) {
  int id = blockIdx.x * blockDim.x + threadIdx.x;
  if (id < n) dst[id] = (_Float16)src[id];
}

// W is [Kin][Nout] row-major (as in reference: x @ W). Store transposed f16
// Wt[Nout][Kin] so B-fragment loads stream along K.
__global__ void wt_transpose_kernel(const float* __restrict__ W,
                                    _Float16* __restrict__ Wt,
                                    int Kin, int Nout) {
  int id = blockIdx.x * blockDim.x + threadIdx.x;
  if (id >= Kin * Nout) return;
  int k = id % Kin;
  int n = id / Kin;
  Wt[(size_t)n * Kin + k] = (_Float16)W[(size_t)k * Nout + n];
}

// ---------------- top-K over overlaps (one wave per (b,i) row) --------------

__global__ void topk_kernel(const float* __restrict__ ious,
                            const float* __restrict__ gmask,
                            int*   __restrict__ oIdx,
                            float* __restrict__ oOv,   // iou_g * mk
                            float* __restrict__ oWf,   // neg_i * mk (wave factor)
                            float* __restrict__ oMk) { // mk
  __shared__ float sv[4 * 320];
  __shared__ int   si[4 * 320];
  int lane = threadIdx.x & 31;
  int wv   = threadIdx.x >> 5;           // 4 waves per block
  int row  = blockIdx.x * 4 + wv;        // < 7200 by construction
  int b = row / NN;
  int i = row - b * NN;
  float negi = 1.0f - gmask[row];
  const float* prow = ious + (size_t)b * (NN * NN) + (size_t)i * NN;

  float tv[10]; int ti[10];
#pragma unroll
  for (int q = 0; q < 10; ++q) { tv[q] = -1.0f; ti[q] = 0; }

  for (int j = lane; j < NN; j += 32) {
    float v = prow[j] * gmask[b * NN + j] * negi;
    int jj = j;
    // sorted-descending insert with static indexing (no scratch spill)
#pragma unroll
    for (int q = 0; q < 10; ++q) {
      if (v > tv[q]) {
        float t = tv[q]; tv[q] = v; v = t;
        int t2 = ti[q]; ti[q] = jj; jj = t2;
      }
    }
  }
  int base = wv * 320 + lane * 10;
#pragma unroll
  for (int q = 0; q < 10; ++q) { sv[base + q] = tv[q]; si[base + q] = ti[q]; }
  __syncthreads();

  if (lane == 0) {
    int wb = wv * 320;
    for (int k = 0; k < KK; ++k) {
      float best = -2.0f; int bp = wb;
      for (int t = 0; t < 320; ++t) {
        float s = sv[wb + t];
        if (s > best) { best = s; bp = wb + t; }
      }
      int bj = si[bp];
      sv[bp] = -3.0f;
      float nmk = gmask[b * NN + bj];
      float mk  = (best >= 0.4f) ? nmk : 0.0f;
      size_t o = (size_t)row * KK + k;
      oIdx[o] = bj;
      oOv[o]  = best * mk;
      oWf[o]  = negi * mk;
      oMk[o]  = mk;
    }
  }
}

// ---------------- feature build: [64 iou] ++ [512 sin/cos] ------------------

__global__ void build_features_kernel(const float* __restrict__ bb,
                                      const int*   __restrict__ tkIdx,
                                      const float* __restrict__ tkOv,
                                      const float* __restrict__ tkWf,
                                      _Float16* __restrict__ featA,
                                      int r0) {
  int r  = r0 + blockIdx.x;      // global feature row
  int ri = r / KK;               // (b,i) row
  int b  = ri / NN;
  int j  = tkIdx[r];
  const float* own = bb + (size_t)ri * 4;
  const float* nb  = bb + ((size_t)b * NN + j) * 4;
  float ocx = 0.5f * (own[0] + own[2]), ocy = 0.5f * (own[1] + own[3]);
  float ow  = own[2] - own[0],          oh  = own[3] - own[1];
  float ncx = 0.5f * (nb[0] + nb[2]),   ncy = 0.5f * (nb[1] + nb[3]);
  float nw  = nb[2] - nb[0],            nh  = nb[3] - nb[1];
  float pmat[4];
  pmat[0] = logf(fmaxf(fabsf(ncx - ocx), 1e-7f));
  pmat[1] = logf(fmaxf(fabsf(ncy - ocy), 1e-7f));
  pmat[2] = logf(fmaxf(fabsf(nw - ow),   1e-7f));
  pmat[3] = logf(fmaxf(fabsf(nh - oh),   1e-7f));
  float ov = tkOv[r], wf = tkWf[r];
  _Float16* orow = featA + (size_t)blockIdx.x * FEATC;
  // dim_t[d] = 10000^((d>>1)/64)  ->  1/dim_t = exp2(-(log2(1e4)/64)*(d>>1))
  const float c_exp = -0.20762050594046382f;   // -log2(10000)/64
  for (int c = threadIdx.x; c < FEATC; c += 192) {
    float val;
    if (c < 64) {
      val = ov;
    } else {
      int e = c - 64;
      int a = e >> 8;            // component pair 0/1
      int s = (e >> 7) & 1;      // 0=sin, 1=cos
      int d = e & 127;           // frequency index
      float invf = exp2f(c_exp * (float)(d >> 1));
      float ang  = 6.283185307179586f * pmat[2 * a + s] * invf;
      val = (s ? cosf(ang) : sinf(ang)) * wf;
    }
    orow[c] = (_Float16)val;
  }
}

// ---------------- WMMA GEMM: one wave per 16(M) x 64(N) tile ----------------
// A: [M][K] f16 row-major. Bt: [N][K] f16 row-major (i.e. B column-major).
// flags: bit0 = relu, bit1 = store f16 (else f32). Bias always present.
// Requires M%16==0, N%64==0, K%32==0.

__device__ __forceinline__ v16h frag_a(const _Float16* p) {
  // 16-bit A layout: lane-half h holds K {8h..8h+7, 16+8h..16+8h+7}
  v8h lo = *(const v8h*)(p);
  v8h hi = *(const v8h*)(p + 16);
  v16h r;
#pragma unroll
  for (int q = 0; q < 8; ++q) { r[q] = lo[q]; r[q + 8] = hi[q]; }
  return r;
}

__device__ __forceinline__ v16h frag_b(const _Float16* p) {
  // 16-bit B layout: lane-half h holds K {8h..8h+7, 16+8h..16+8h+7}
  v8h lo = *(const v8h*)(p);
  v8h hi = *(const v8h*)(p + 16);
  v16h r;
#pragma unroll
  for (int q = 0; q < 8; ++q) { r[q] = lo[q]; r[q + 8] = hi[q]; }
  return r;
}

__global__ void gemm16x64_kernel(const _Float16* __restrict__ A,
                                 const _Float16* __restrict__ Bt,
                                 const float* __restrict__ bias,
                                 void* __restrict__ out,
                                 int M, int N, int K, int flags) {
  int tilesN = N >> 6;                 // 64-wide N tiles
  int tm = blockIdx.x / tilesN;
  int tn = blockIdx.x % tilesN;
  int lane = threadIdx.x;              // 32 threads = 1 wave, EXEC all ones
  int half = lane >> 4;
  int r16  = lane & 15;
  const _Float16* Ap = A  + (size_t)(tm * 16 + r16) * K + half * 8;
  const _Float16* Bp = Bt + (size_t)(tn * 64 + r16) * K + half * 8;
  size_t bs = (size_t)16 * K;          // stride between 16-col B groups
  v8f a0 = {0.f,0.f,0.f,0.f,0.f,0.f,0.f,0.f};
  v8f a1 = a0, a2 = a0, a3 = a0;
  for (int k = 0; k < K; k += 32) {
    v16h af = frag_a(Ap + k);
    v16h b0 = frag_b(Bp + k);
    v16h b1 = frag_b(Bp + bs + k);
    v16h b2 = frag_b(Bp + 2 * bs + k);
    v16h b3 = frag_b(Bp + 3 * bs + k);
    a0 = __builtin_amdgcn_wmma_f32_16x16x32_f16(false, af, false, b0, (short)0, a0, false, false);
    a1 = __builtin_amdgcn_wmma_f32_16x16x32_f16(false, af, false, b1, (short)0, a1, false, false);
    a2 = __builtin_amdgcn_wmma_f32_16x16x32_f16(false, af, false, b2, (short)0, a2, false, false);
    a3 = __builtin_amdgcn_wmma_f32_16x16x32_f16(false, af, false, b3, (short)0, a3, false, false);
  }
#define STORE_TILE(ACC, SUB)                                                   \
  {                                                                            \
    int n = tn * 64 + (SUB) * 16 + r16;                                        \
    float bv = bias[n];                                                        \
    _Pragma("unroll")                                                          \
    for (int v = 0; v < 8; ++v) {                                              \
      int m = tm * 16 + half * 8 + v;                                          \
      float x = ACC[v] + bv;                                                   \
      if (flags & 1) x = fmaxf(x, 0.0f);                                       \
      if (flags & 2) ((_Float16*)out)[(size_t)m * N + n] = (_Float16)x;        \
      else           ((float*)out)[(size_t)m * N + n] = x;                     \
    }                                                                          \
  }
  STORE_TILE(a0, 0)
  STORE_TILE(a1, 1)
  STORE_TILE(a2, 2)
  STORE_TILE(a3, 3)
#undef STORE_TILE
}

// ---------------- masked max over K, combine, final scale -------------------

__global__ void maxagg_kernel(const float* __restrict__ feat,   // [nbi*10][256]
                              const float* __restrict__ mk,     // [nbi*10]
                              float* __restrict__ agg,          // [nbi][256]
                              int nbi) {
  int g = blockIdx.x;
  int n = threadIdx.x;
  if (g >= nbi) return;
  float m = -3.4e38f;
  for (int k = 0; k < KK; ++k) {
    float v = feat[((size_t)g * KK + k) * DD + n] * mk[g * KK + k];
    m = fmaxf(m, v);
  }
  agg[(size_t)g * DD + n] = m;
}

__global__ void combine_kernel(const float* __restrict__ cur,
                               const float* __restrict__ agg,
                               const float* __restrict__ gmask,
                               _Float16* __restrict__ ctH) {
  int id = blockIdx.x * blockDim.x + threadIdx.x;   // 7200*256
  int ri = id >> 8;
  float negi = 1.0f - gmask[ri];
  ctH[id] = (_Float16)(cur[id] * negi + agg[id]);
}

__global__ void scale_neg_kernel(float* __restrict__ out,
                                 const float* __restrict__ gmask) {
  int id = blockIdx.x * blockDim.x + threadIdx.x;
  int ri = id >> 8;
  out[id] *= (1.0f - gmask[ri]);
}

// ---------------------------------------------------------------------------

extern "C" void kernel_launch(void* const* d_in, const int* in_sizes, int n_in,
                              void* d_out, int out_size, void* d_ws, size_t ws_size,
                              hipStream_t stream) {
  (void)in_sizes; (void)n_in; (void)out_size; (void)ws_size;
  const float* tgt    = (const float*)d_in[0];
  const float* ious   = (const float*)d_in[1];
  const float* bboxes = (const float*)d_in[2];
  const float* gmask  = (const float*)d_in[3];
  const float* W1 = (const float*)d_in[4];  const float* b1 = (const float*)d_in[5];
  const float* W2 = (const float*)d_in[6];  const float* b2 = (const float*)d_in[7];
  const float* W3 = (const float*)d_in[8];  const float* b3 = (const float*)d_in[9];
  const float* W4 = (const float*)d_in[10]; const float* b4 = (const float*)d_in[11];
  const float* W5 = (const float*)d_in[12]; const float* b5 = (const float*)d_in[13];

  // workspace carve-out (256B aligned), ~49 MB total
  char* w = (char*)d_ws;
  auto alloc = [&](size_t bytes) -> void* {
    void* p = (void*)w;
    w += (bytes + 255) & ~(size_t)255;
    return p;
  };
  _Float16* tgtH  = (_Float16*)alloc((size_t)BSN * DD * 2);
  _Float16* W1t   = (_Float16*)alloc((size_t)DD * DD * 2);
  _Float16* W2t   = (_Float16*)alloc((size_t)DD * DD * 2);
  _Float16* W3t   = (_Float16*)alloc((size_t)FEATC * DD * 2);
  _Float16* W4t   = (_Float16*)alloc((size_t)DD * DD * 2);
  _Float16* W5t   = (_Float16*)alloc((size_t)DD * DD * 2);
  int*      tkIdx = (int*)alloc((size_t)FROWS * 4);
  float*    tkOv  = (float*)alloc((size_t)FROWS * 4);
  float*    tkWf  = (float*)alloc((size_t)FROWS * 4);
  float*    tkMk  = (float*)alloc((size_t)FROWS * 4);
  _Float16* featA = (_Float16*)alloc((size_t)CH_R * FEATC * 2);
  _Float16* hidden= (_Float16*)alloc((size_t)CH_R * DD * 2);
  float*    feat  = (float*)alloc((size_t)CH_R * DD * 4);
  float*    aggB  = (float*)alloc((size_t)BSN * DD * 4);
  float*    curB  = (float*)alloc((size_t)BSN * DD * 4);
  _Float16* curH  = (_Float16*)alloc((size_t)BSN * DD * 2);
  _Float16* ctH   = (_Float16*)alloc((size_t)BSN * DD * 2);

  // 1) weight / activation conversion to f16 (weights transposed to [N][K])
  wt_transpose_kernel<<<(DD * DD + 255) / 256, 256, 0, stream>>>(W1, W1t, DD, DD);
  wt_transpose_kernel<<<(DD * DD + 255) / 256, 256, 0, stream>>>(W2, W2t, DD, DD);
  wt_transpose_kernel<<<(FEATC * DD + 255) / 256, 256, 0, stream>>>(W3, W3t, FEATC, DD);
  wt_transpose_kernel<<<(DD * DD + 255) / 256, 256, 0, stream>>>(W4, W4t, DD, DD);
  wt_transpose_kernel<<<(DD * DD + 255) / 256, 256, 0, stream>>>(W5, W5t, DD, DD);
  cvt_f16_kernel<<<(BSN * DD + 255) / 256, 256, 0, stream>>>(tgt, tgtH, BSN * DD);

  // 2) top-K neighbor selection (one wave per row, 4 waves/block)
  topk_kernel<<<BSN / 4, 128, 0, stream>>>(ious, gmask, tkIdx, tkOv, tkWf, tkMk);

  // 3) cur = relu(tgt@W1+b1)@W2+b2   (WMMA, 16x64 tiles)
  gemm16x64_kernel<<<(BSN / 16) * (DD / 64), 32, 0, stream>>>(tgtH, W1t, b1, curH,
                                                              BSN, DD, DD, /*relu|f16*/3);
  gemm16x64_kernel<<<(BSN / 16) * (DD / 64), 32, 0, stream>>>(curH, W2t, b2, curB,
                                                              BSN, DD, DD, 0);

  // 4) chunked feature path: build -> W3 GEMM(relu,f16) -> W4 GEMM(f32) -> masked max
  for (int ch = 0; ch < FROWS / CH_R; ++ch) {
    int r0 = ch * CH_R;
    build_features_kernel<<<CH_R, 192, 0, stream>>>(bboxes, tkIdx, tkOv, tkWf,
                                                    featA, r0);
    gemm16x64_kernel<<<(CH_R / 16) * (DD / 64), 32, 0, stream>>>(featA, W3t, b3, hidden,
                                                                 CH_R, DD, FEATC, 3);
    gemm16x64_kernel<<<(CH_R / 16) * (DD / 64), 32, 0, stream>>>(hidden, W4t, b4, feat,
                                                                 CH_R, DD, DD, 0);
    maxagg_kernel<<<CH_BI, DD, 0, stream>>>(feat, tkMk + r0,
                                            aggB + (size_t)ch * CH_BI * DD, CH_BI);
  }

  // 5) cur_tgt = cur*neg + agg ; out = relu(cur_tgt@W5+b5) * neg
  combine_kernel<<<BSN, DD, 0, stream>>>(curB, aggB, gmask, ctH);
  gemm16x64_kernel<<<(BSN / 16) * (DD / 64), 32, 0, stream>>>(ctH, W5t, b5,
                                                              (float*)d_out,
                                                              BSN, DD, DD, /*relu*/1);
  scale_neg_kernel<<<BSN, DD, 0, stream>>>((float*)d_out, gmask);
}
